// ProtGNN_37374805410154
// MI455X (gfx1250) — matrix operand order repaired and checked by reading, not compiled, not run on hardware
//
#include <hip/hip_runtime.h>

#define N_NODES 100000
#define N_EDGES 1600000
#define N_GRAPHS 200
#define IN_DIM 25
#define HID 256
#define OUT_DIM 128

typedef __attribute__((ext_vector_type(16))) _Float16 v16h;
typedef __attribute__((ext_vector_type(8)))  _Float16 v8h;
typedef __attribute__((ext_vector_type(8)))  float    v8f;

// ---------------- utility kernels ----------------

__global__ void fill_zero(float* __restrict__ p, long n) {
    long i = (long)blockIdx.x * blockDim.x + threadIdx.x;
    if (i < n) p[i] = 0.f;
}

__global__ void edge_count(const long long* __restrict__ dst, int E,
                           float* __restrict__ cnt) {
    int i = blockIdx.x * blockDim.x + threadIdx.x;
    if (i < E) atomicAdd(&cnt[(int)dst[i]], 1.f);
}

__global__ void inv_count(float* __restrict__ cnt, int n) {
    int i = blockIdx.x * blockDim.x + threadIdx.x;
    if (i < n) cnt[i] = 1.f / fmaxf(cnt[i], 1.f);
}

// scalar scatter-add: one thread per (edge, channel) -- used for d=25
__global__ void scatter_add1(const long long* __restrict__ src,
                             const long long* __restrict__ dst, int E,
                             const float* __restrict__ x, int d,
                             float* __restrict__ agg) {
    long idx = (long)blockIdx.x * blockDim.x + threadIdx.x;
    long total = (long)E * d;
    if (idx >= total) return;
    int e = (int)(idx / d);
    int c = (int)(idx % d);
    atomicAdd(&agg[(long)dst[e] * d + c], x[(long)src[e] * d + c]);
}

// float4 scatter-add: one thread per (edge, 4-channel group) -- d % 4 == 0
__global__ void scatter_add4(const long long* __restrict__ src,
                             const long long* __restrict__ dst, int E,
                             const float* __restrict__ x, int d4, int d,
                             float* __restrict__ agg) {
    long idx = (long)blockIdx.x * blockDim.x + threadIdx.x;
    long total = (long)E * d4;
    if (idx >= total) return;
    int e  = (int)(idx / d4);
    int cg = (int)(idx % d4);
    const float4 v = ((const float4*)(x + (long)src[e] * d))[cg];
    float* a = agg + (long)dst[e] * d + (long)cg * 4;
    atomicAdd(a + 0, v.x);
    atomicAdd(a + 1, v.y);
    atomicAdd(a + 2, v.z);
    atomicAdd(a + 3, v.w);
}

__global__ void scale_rows(float* __restrict__ agg, const float* __restrict__ inv,
                           int n, int d) {
    long idx = (long)blockIdx.x * blockDim.x + threadIdx.x;
    if (idx < (long)n * d) agg[idx] *= inv[idx / d];
}

// segment-max pooling; inputs are post-ReLU (>= 0) so uint bit-pattern max == float max
__global__ void pool_max(const float* __restrict__ h,
                         const long long* __restrict__ batch,
                         int n, int d, float* __restrict__ pool,
                         int poolLd, int colOff) {
    long idx = (long)blockIdx.x * blockDim.x + threadIdx.x;
    if (idx >= (long)n * d) return;
    int i = (int)(idx / d);
    int c = (int)(idx % d);
    unsigned int v = __float_as_uint(h[idx]);
    atomicMax((unsigned int*)&pool[(long)batch[i] * poolLd + colOff + c], v);
}

// ---------------- f16 packing kernels (hoist all guards out of the GEMM) ----

// f32 [M,K] row-major -> f16 [Mpad,Kpad] row-major, zero-filled
__global__ void pack_a_f16(const float* __restrict__ src, _Float16* __restrict__ dst,
                           int M, int K, int Mpad, int Kpad) {
    long idx = (long)blockIdx.x * blockDim.x + threadIdx.x;
    long total = (long)Mpad * Kpad;
    if (idx >= total) return;
    int m = (int)(idx / Kpad);
    int k = (int)(idx % Kpad);
    dst[idx] = (_Float16)((m < M && k < K) ? src[(long)m * K + k] : 0.f);
}

// f32 B [K,N] row-major -> f16 Bt [N,Kpad] (transposed), zero-filled
__global__ void pack_bt_f16(const float* __restrict__ B, _Float16* __restrict__ dst,
                            int K, int N, int Kpad) {
    long idx = (long)blockIdx.x * blockDim.x + threadIdx.x;
    long total = (long)N * Kpad;
    if (idx >= total) return;
    int nn = (int)(idx / Kpad);
    int k  = (int)(idx % Kpad);
    dst[idx] = (_Float16)((k < K) ? B[(long)k * N + nn] : 0.f);
}

// ---------------- WMMA GEMM ----------------
// C[M,N] = relu?( A1@B1 + A2@B2 + bias ), A* are f16 [tilesM*16, K*] (K%32==0),
// B*t are f16 [N, K*] (weight transposed). One wave per 16x16 tile.
// Fragment loads are contiguous b128s: no predication in the hot loop.
__global__ void wmma_gemm2(const _Float16* __restrict__ A1, int K1,
                           const _Float16* __restrict__ B1t,
                           const _Float16* __restrict__ A2, int K2,
                           const _Float16* __restrict__ B2t,
                           const float* __restrict__ bias,
                           float* __restrict__ C, int M, int N, int doRelu) {
    int wid    = blockIdx.x * (blockDim.x >> 5) + (threadIdx.x >> 5);
    int tilesN = N >> 4;
    int tilesM = (M + 15) >> 4;
    if (wid >= tilesM * tilesN) return;   // whole-wave uniform exit
    int tm = wid / tilesN, tn = wid % tilesN;
    int m0 = tm << 4, n0 = tn << 4;

    int lane = threadIdx.x & 31;
    int half = lane >> 4;
    int lm   = lane & 15;
    int m = m0 + lm;
    int n = n0 + lm;

    v8f acc = {};
    for (int pass = 0; pass < 2; ++pass) {
        const _Float16* A  = pass ? A2 : A1;
        const _Float16* Bt = pass ? B2t : B1t;
        int K              = pass ? K2 : K1;
        if (!A) continue;
        // A fragment lane layout: elements 0-7 -> K = kk + half*8 + e,
        //                         elements 8-15 -> K = kk + half*8 + 16 + e
        const _Float16* ap = A + (long)m * K + half * 8;
        // B fragment lane layout: elements 0-7 -> K = kk + half*8 + e,
        //                         elements 8-15 -> K = kk + half*8 + 16 + e (column n)
        const _Float16* bp = Bt + (long)n * K + half * 8;
        for (int kk = 0; kk < K; kk += 32) {
            v8h alo = *(const v8h*)(ap + kk);
            v8h ahi = *(const v8h*)(ap + kk + 16);
            v8h blo = *(const v8h*)(bp + kk);
            v8h bhi = *(const v8h*)(bp + kk + 16);
            v16h a = __builtin_shufflevector(alo, ahi, 0, 1, 2, 3, 4, 5, 6, 7,
                                             8, 9, 10, 11, 12, 13, 14, 15);
            v16h b = __builtin_shufflevector(blo, bhi, 0, 1, 2, 3, 4, 5, 6, 7,
                                             8, 9, 10, 11, 12, 13, 14, 15);
            acc = __builtin_amdgcn_wmma_f32_16x16x32_f16(
                false, a, false, b, (short)0, acc, false, false);
        }
    }

    float bv = bias ? bias[n] : 0.f;
    #pragma unroll
    for (int r = 0; r < 8; ++r) {
        int row = m0 + r + half * 8;   // C/D layout: VGPR r -> M=r / r+8
        if (row < M) {
            float v = acc[r] + bv;
            if (doRelu) v = fmaxf(v, 0.f);
            C[(long)row * N + n] = v;
        }
    }
}

// ---------------- host launch ----------------

static inline int cdiv(long a, long b) { return (int)((a + b - 1) / b); }

extern "C" void kernel_launch(void* const* d_in, const int* in_sizes, int n_in,
                              void* d_out, int out_size, void* d_ws, size_t ws_size,
                              hipStream_t stream) {
    const float*     x     = (const float*)d_in[0];
    const long long* ei    = (const long long*)d_in[1];   // int64 [2, E]
    const long long* batch = (const long long*)d_in[2];   // int64 [N]
    const float* W1l = (const float*)d_in[3];
    const float* W1r = (const float*)d_in[4];
    const float* b1  = (const float*)d_in[5];
    const float* W2l = (const float*)d_in[6];
    const float* W2r = (const float*)d_in[7];
    const float* b2  = (const float*)d_in[8];
    const float* W3l = (const float*)d_in[9];
    const float* W3r = (const float*)d_in[10];
    const float* b3  = (const float*)d_in[11];
    const float* Wlin1 = (const float*)d_in[12];
    const float* blin1 = (const float*)d_in[13];
    const float* Wlin2 = (const float*)d_in[14];
    const float* blin2 = (const float*)d_in[15];

    const long long* srcE = ei;            // edge_index[0]
    const long long* dstE = ei + N_EDGES;  // edge_index[1]

    // ---- workspace carve-up (sequential; every slab stays 16B aligned) ----
    float* wsf  = (float*)d_ws;
    float* cnt  = wsf;  wsf += N_NODES;
    float* agg  = wsf;  wsf += (long)N_NODES * HID;
    float* bufA = wsf;  wsf += (long)N_NODES * HID;   // x1, then x3
    float* bufB = wsf;  wsf += (long)N_NODES * HID;   // x2
    float* pool = wsf;  wsf += (long)N_GRAPHS * 3 * HID;
    float* hbuf = wsf;  wsf += (long)N_GRAPHS * HID;

    _Float16* wsh   = (_Float16*)wsf;
    _Float16* aggH  = wsh;  wsh += (long)N_NODES * HID;   // packed aggregate
    _Float16* xH    = wsh;  wsh += (long)N_NODES * HID;   // packed root features
    _Float16* W1lT  = wsh;  wsh += 256 * 32;
    _Float16* W1rT  = wsh;  wsh += 256 * 32;
    _Float16* W2lT  = wsh;  wsh += 256 * 256;
    _Float16* W2rT  = wsh;  wsh += 256 * 256;
    _Float16* W3lT  = wsh;  wsh += 256 * 256;
    _Float16* W3rT  = wsh;  wsh += 256 * 256;
    _Float16* Wl1T  = wsh;  wsh += 256 * 768;
    _Float16* Wl2T  = wsh;  wsh += 128 * 256;
    _Float16* poolH = wsh;  wsh += 208 * 768;
    _Float16* hH    = wsh;  wsh += 208 * 256;

    float* out = (float*)d_out;

    const int T = 256;                 // 8 wave32 per block
    const int WPB = T / 32;

    // ---- weight packing (f16, transposed) ----
    pack_bt_f16<<<cdiv(256 * 32, T), T, 0, stream>>>(W1l, W1lT, IN_DIM, HID, 32);
    pack_bt_f16<<<cdiv(256 * 32, T), T, 0, stream>>>(W1r, W1rT, IN_DIM, HID, 32);
    pack_bt_f16<<<cdiv(256 * 256, T), T, 0, stream>>>(W2l, W2lT, HID, HID, HID);
    pack_bt_f16<<<cdiv(256 * 256, T), T, 0, stream>>>(W2r, W2rT, HID, HID, HID);
    pack_bt_f16<<<cdiv(256 * 256, T), T, 0, stream>>>(W3l, W3lT, HID, HID, HID);
    pack_bt_f16<<<cdiv(256 * 256, T), T, 0, stream>>>(W3r, W3rT, HID, HID, HID);
    pack_bt_f16<<<cdiv(256 * 768, T), T, 0, stream>>>(Wlin1, Wl1T, 3 * HID, HID, 3 * HID);
    pack_bt_f16<<<cdiv(128 * 256, T), T, 0, stream>>>(Wlin2, Wl2T, HID, OUT_DIM, HID);

    // ---- degree counts -> 1/max(cnt,1); zero pool ----
    fill_zero<<<cdiv(N_NODES, T), T, 0, stream>>>(cnt, N_NODES);
    edge_count<<<cdiv(N_EDGES, T), T, 0, stream>>>(dstE, N_EDGES, cnt);
    inv_count<<<cdiv(N_NODES, T), T, 0, stream>>>(cnt, N_NODES);
    fill_zero<<<cdiv((long)N_GRAPHS * 3 * HID, T), T, 0, stream>>>(pool, (long)N_GRAPHS * 3 * HID);

    // ---- layer 1 (d = 25, Kpad = 32) ----
    fill_zero<<<cdiv((long)N_NODES * IN_DIM, T), T, 0, stream>>>(agg, (long)N_NODES * IN_DIM);
    scatter_add1<<<cdiv((long)N_EDGES * IN_DIM, T), T, 0, stream>>>(srcE, dstE, N_EDGES, x, IN_DIM, agg);
    scale_rows<<<cdiv((long)N_NODES * IN_DIM, T), T, 0, stream>>>(agg, cnt, N_NODES, IN_DIM);
    pack_a_f16<<<cdiv((long)N_NODES * 32, T), T, 0, stream>>>(agg, aggH, N_NODES, IN_DIM, N_NODES, 32);
    pack_a_f16<<<cdiv((long)N_NODES * 32, T), T, 0, stream>>>(x, xH, N_NODES, IN_DIM, N_NODES, 32);
    {
        int tiles = (N_NODES / 16) * (HID / 16);
        wmma_gemm2<<<cdiv(tiles, WPB), T, 0, stream>>>(aggH, 32, W1lT, xH, 32, W1rT,
                                                       b1, bufA, N_NODES, HID, 1);
    }
    pool_max<<<cdiv((long)N_NODES * HID, T), T, 0, stream>>>(bufA, batch, N_NODES, HID, pool, 3 * HID, 0);

    // ---- layer 2 (d = 256) ----
    fill_zero<<<cdiv((long)N_NODES * HID, T), T, 0, stream>>>(agg, (long)N_NODES * HID);
    scatter_add4<<<cdiv((long)N_EDGES * (HID / 4), T), T, 0, stream>>>(srcE, dstE, N_EDGES, bufA, HID / 4, HID, agg);
    scale_rows<<<cdiv((long)N_NODES * HID, T), T, 0, stream>>>(agg, cnt, N_NODES, HID);
    pack_a_f16<<<cdiv((long)N_NODES * HID, T), T, 0, stream>>>(agg, aggH, N_NODES, HID, N_NODES, HID);
    pack_a_f16<<<cdiv((long)N_NODES * HID, T), T, 0, stream>>>(bufA, xH, N_NODES, HID, N_NODES, HID);
    {
        int tiles = (N_NODES / 16) * (HID / 16);
        wmma_gemm2<<<cdiv(tiles, WPB), T, 0, stream>>>(aggH, HID, W2lT, xH, HID, W2rT,
                                                       b2, bufB, N_NODES, HID, 1);
    }
    pool_max<<<cdiv((long)N_NODES * HID, T), T, 0, stream>>>(bufB, batch, N_NODES, HID, pool, 3 * HID, HID);

    // ---- layer 3 (d = 256) ----
    fill_zero<<<cdiv((long)N_NODES * HID, T), T, 0, stream>>>(agg, (long)N_NODES * HID);
    scatter_add4<<<cdiv((long)N_EDGES * (HID / 4), T), T, 0, stream>>>(srcE, dstE, N_EDGES, bufB, HID / 4, HID, agg);
    scale_rows<<<cdiv((long)N_NODES * HID, T), T, 0, stream>>>(agg, cnt, N_NODES, HID);
    pack_a_f16<<<cdiv((long)N_NODES * HID, T), T, 0, stream>>>(agg, aggH, N_NODES, HID, N_NODES, HID);
    pack_a_f16<<<cdiv((long)N_NODES * HID, T), T, 0, stream>>>(bufB, xH, N_NODES, HID, N_NODES, HID);
    {
        int tiles = (N_NODES / 16) * (HID / 16);
        wmma_gemm2<<<cdiv(tiles, WPB), T, 0, stream>>>(aggH, HID, W3lT, xH, HID, W3rT,
                                                       b3, bufA, N_NODES, HID, 1);
    }
    pool_max<<<cdiv((long)N_NODES * HID, T), T, 0, stream>>>(bufA, batch, N_NODES, HID, pool, 3 * HID, 2 * HID);

    // ---- MLP head (M=200 padded to 208 rows in the f16 pack) ----
    pack_a_f16<<<cdiv(208L * 768, T), T, 0, stream>>>(pool, poolH, N_GRAPHS, 3 * HID, 208, 3 * HID);
    {
        int tiles = cdiv(N_GRAPHS, 16) * (HID / 16);
        wmma_gemm2<<<cdiv(tiles, WPB), T, 0, stream>>>(poolH, 3 * HID, Wl1T,
                                                       nullptr, 0, nullptr,
                                                       blin1, hbuf, N_GRAPHS, HID, 1);
    }
    pack_a_f16<<<cdiv(208L * 256, T), T, 0, stream>>>(hbuf, hH, N_GRAPHS, HID, 208, HID);
    {
        int tiles = cdiv(N_GRAPHS, 16) * (OUT_DIM / 16);
        wmma_gemm2<<<cdiv(tiles, WPB), T, 0, stream>>>(hH, HID, Wl2T,
                                                       nullptr, 0, nullptr,
                                                       blin2, out, N_GRAPHS, OUT_DIM, 0);
    }
}
